// ConvexSimilarityHash_35777077575837
// MI455X (gfx1250) — matrix operand order repaired and checked by reading, not compiled, not run on hardware
//
#include <hip/hip_runtime.h>
#include <hip/hip_bf16.h>
#include <math.h>

// ---------------- problem constants ----------------
#define B_   16
#define T_   4096
#define E_   1024
#define WIN_ 15
#define HALF_ 2048
#define EPS_ 1e-6f
#define PI_  3.14159265358979323846f
#define KCH_ 8          // split-K chunks per output tile

typedef __attribute__((ext_vector_type(16))) __bf16 v16bf;
typedef __attribute__((ext_vector_type(8)))  float  v8f;

// ---------------------------------------------------------------------------
// Kernel 1: per-(b,t) row statistics. One wave per row of x.
//   c0[row]     = x[row][0] * taper[t]
//   phases[row] = (c0>=0 ? 0 : pi) + sum_{e>=1} asin(clip(x[row][e]*taper[t]/(|c0|+eps)))
// ---------------------------------------------------------------------------
__global__ void csh_rowstats(const float* __restrict__ x,
                             const float* __restrict__ taper,
                             float* __restrict__ phases,
                             float* __restrict__ c0out) {
    const int wave = (blockIdx.x * blockDim.x + threadIdx.x) >> 5;   // row in [0, B*T)
    const int lane = threadIdx.x & 31;
    const int t = wave & (T_ - 1);

    const float* __restrict__ row = x + (size_t)wave * E_;
    const float tap = taper[t];
    const float x0  = row[0] * tap;                 // broadcast load
    const float inv = tap / (fabsf(x0) + EPS_);     // fold taper into the divide

    float s = 0.f;
#pragma unroll 4
    for (int i = 0; i < 32; ++i) {
        const int e = 1 + lane + (i << 5);
        if (e < E_) {
            float v = row[e] * inv;
            v = fminf(fmaxf(v, -1.f + EPS_), 1.f - EPS_);
            s += asinf(v);
        }
    }
    for (int off = 16; off > 0; off >>= 1)          // wave32 butterfly
        s += __shfl_xor(s, off, 32);

    if (lane == 0) {
        phases[wave] = ((x0 >= 0.f) ? 0.f : PI_) + s;
        c0out[wave]  = x0;
    }
}

// ---------------------------------------------------------------------------
// Kernel 2: causal triangular moving average over previous WIN steps of c0.
// ---------------------------------------------------------------------------
__global__ void csh_tdiff(const float* __restrict__ c0,
                          float* __restrict__ td) {
    const int idx = blockIdx.x * blockDim.x + threadIdx.x;   // [0, B*T)
    const int t = idx & (T_ - 1);
    float num = 0.f;
#pragma unroll
    for (int d = 1; d <= WIN_; ++d)
        if (d <= t) num += (float)d * c0[idx - d];
    const int   sw   = t < WIN_ ? t : WIN_;
    const float norm = 0.5f * (float)sw * (float)(sw + 1);
    td[idx] = (norm > 0.f) ? num / norm : 0.f;
}

// ---------------------------------------------------------------------------
// Kernel 3: split-K M=16 GEMM partial via v_wmma_f32_16x16x32_bf16.
//   partial[tile,chunk][m,n] = sum_{k in chunk} A[m,k] * W[tile*16+n, k]
// One wave = one (N-tile, K-chunk). Two independent accumulator chains
// (even/odd 32-blocks) overlap WMMA with the load/convert stream.
//
// Fragment layouts as loaded here (wave32; A and B share one k order):
//   A 16x32 bf16 : lane L, M=L&15, elem j -> K=(L<16?0:16)+j  (contiguous)
//   B 32x16 bf16 : lane L, N=L&15, elem j -> K=(L<16?0:16)+j  (contiguous)
//   C/D 16x16 f32: VGPR r, lane L -> M = r + 8*(L>=16), N = L&15
// ---------------------------------------------------------------------------
__device__ __forceinline__ v16bf csh_afrag(const float* __restrict__ p) {
    // p points at A-row + k-block + koff ; needs 16 consecutive floats
    const float4 a0 = *(const float4*)(p + 0);
    const float4 a1 = *(const float4*)(p + 4);
    const float4 a2 = *(const float4*)(p + 8);
    const float4 a3 = *(const float4*)(p + 12);
    v16bf a;
    a[0]=(__bf16)a0.x; a[1]=(__bf16)a0.y; a[2]=(__bf16)a0.z; a[3]=(__bf16)a0.w;
    a[4]=(__bf16)a1.x; a[5]=(__bf16)a1.y; a[6]=(__bf16)a1.z; a[7]=(__bf16)a1.w;
    a[8]=(__bf16)a2.x; a[9]=(__bf16)a2.y; a[10]=(__bf16)a2.z; a[11]=(__bf16)a2.w;
    a[12]=(__bf16)a3.x; a[13]=(__bf16)a3.y; a[14]=(__bf16)a3.z; a[15]=(__bf16)a3.w;
    return a;
}

__device__ __forceinline__ v16bf csh_bfrag(const float* __restrict__ p) {
    // p points at W-row + k-block + koff ; needs 16 consecutive floats
    const float4 b0 = *(const float4*)(p + 0);
    const float4 b1 = *(const float4*)(p + 4);
    const float4 b2 = *(const float4*)(p + 8);
    const float4 b3 = *(const float4*)(p + 12);
    v16bf b;
    b[0]=(__bf16)b0.x; b[1]=(__bf16)b0.y; b[2]=(__bf16)b0.z; b[3]=(__bf16)b0.w;
    b[4]=(__bf16)b1.x; b[5]=(__bf16)b1.y; b[6]=(__bf16)b1.z; b[7]=(__bf16)b1.w;
    b[8]=(__bf16)b2.x; b[9]=(__bf16)b2.y; b[10]=(__bf16)b2.z; b[11]=(__bf16)b2.w;
    b[12]=(__bf16)b3.x; b[13]=(__bf16)b3.y; b[14]=(__bf16)b3.z; b[15]=(__bf16)b3.w;
    return b;
}

__global__ void csh_gemm16_partial(const float* __restrict__ A,     // 16 x Kdim
                                   const float* __restrict__ W,     // Ncols x Kdim
                                   float* __restrict__ partials,    // tiles*KCH_*256
                                   int Kdim, int Ncols) {
    const int wave  = (blockIdx.x * blockDim.x + threadIdx.x) >> 5;
    const int tile  = wave / KCH_;           // N-tile id
    const int chunk = wave - tile * KCH_;    // K-chunk id
    const int lane  = threadIdx.x & 31;
    const int mn    = lane & 15;             // M for A-frag, N for B-frag
    const int hi    = lane >> 4;

    const int klen = Kdim / KCH_;            // 512 (layer1) / 256 (layer2)
    const int k0   = chunk * klen;

    const float* __restrict__ Arow = A + (size_t)mn * Kdim + k0 + (hi << 4);
    const float* __restrict__ Wrow = W + (size_t)(tile * 16 + mn) * Kdim + k0 + (hi << 4);

    v8f acc0 = {}, acc1 = {};
    for (int k = 0; k < klen; k += 64) {     // two independent WMMA chains
        const v16bf a0 = csh_afrag(Arow + k);
        const v16bf b0 = csh_bfrag(Wrow + k);
        const v16bf a1 = csh_afrag(Arow + k + 32);
        const v16bf b1 = csh_bfrag(Wrow + k + 32);
        __builtin_prefetch(Wrow + k + 512, 0, 1);   // global_prefetch_b8
        acc0 = __builtin_amdgcn_wmma_f32_16x16x32_bf16(
                   false, a0, false, b0, (short)0, acc0, false, false);
        acc1 = __builtin_amdgcn_wmma_f32_16x16x32_bf16(
                   false, a1, false, b1, (short)0, acc1, false, false);
    }

    float* __restrict__ out = partials + (size_t)(tile * KCH_ + chunk) * 256;
#pragma unroll
    for (int r = 0; r < 8; ++r) {
        const int m = r + (hi << 3);
        out[m * 16 + mn] = acc0[r] + acc1[r];
    }
}

// ---------------------------------------------------------------------------
// Kernel 4: reduce split-K partials + bias + activation.
// One thread per output element (16 * Ncols). Deterministic (no atomics).
// act: 0 = SiLU, 1 = tanh. Writes out[(m*Ncols + n)*ostride + ooff].
// ---------------------------------------------------------------------------
__global__ void csh_reduce_act(const float* __restrict__ partials,
                               const float* __restrict__ bias,
                               float* __restrict__ out,
                               int Ncols, int ostride, int ooff, int act) {
    const int idx = blockIdx.x * blockDim.x + threadIdx.x;   // [0, 16*Ncols)
    const int m = idx / Ncols;
    const int n = idx - m * Ncols;
    const int tile = n >> 4;
    const int nl   = n & 15;

    const float* __restrict__ p = partials + (size_t)tile * KCH_ * 256 + m * 16 + nl;
    float v = bias[n];
#pragma unroll
    for (int c = 0; c < KCH_; ++c)
        v += p[c * 256];

    v = (act == 0) ? (v / (1.f + __expf(-v)))   // SiLU
                   : tanhf(v);                  // tanh
    out[((size_t)m * Ncols + n) * ostride + ooff] = v;
}

// ---------------------------------------------------------------------------
// Launcher. setup_inputs order:
//   0:x 1:taper 2:cW1 3:cb1 4:cW2 5:cb2 6:pW1 7:pb1 8:pW2 9:pb2
// Output: (B, HALF, 2) fp32, interleaved [thash, tdiff].
// Workspace (floats):
//   phases[65536] | c0[65536] | td[65536] | h1c[32768] | h1p[32768]
//   | partials[128*KCH_*256 = 262144]   (reused sequentially by all 4 GEMMs)
// ---------------------------------------------------------------------------
extern "C" void kernel_launch(void* const* d_in, const int* in_sizes, int n_in,
                              void* d_out, int out_size, void* d_ws, size_t ws_size,
                              hipStream_t stream) {
    const float* x     = (const float*)d_in[0];
    const float* taper = (const float*)d_in[1];
    const float* cW1   = (const float*)d_in[2];
    const float* cb1   = (const float*)d_in[3];
    const float* cW2   = (const float*)d_in[4];
    const float* cb2   = (const float*)d_in[5];
    const float* pW1   = (const float*)d_in[6];
    const float* pb1   = (const float*)d_in[7];
    const float* pW2   = (const float*)d_in[8];
    const float* pb2   = (const float*)d_in[9];
    float* out = (float*)d_out;
    float* ws  = (float*)d_ws;

    const int BT = B_ * T_;                  // 65536
    float* phases = ws;                      // B*T
    float* c0     = ws + BT;                 // B*T
    float* td     = ws + 2 * BT;             // B*T
    float* h1c    = ws + 3 * BT;             // B*HALF
    float* h1p    = h1c + B_ * HALF_;        // B*HALF
    float* part   = h1p + B_ * HALF_;        // 128*KCH_*256 floats (1 MB)

    const int tiles      = HALF_ / 16;               // 128
    const int gemmBlocks = tiles * KCH_ / 8;         // 1024 waves -> 128 blocks
    const int redBlocks  = (B_ * HALF_) / 256;       // 32768 threads -> 128 blocks

    // 1) row statistics: one wave per (b,t) row
    csh_rowstats<<<BT / 8, 256, 0, stream>>>(x, taper, phases, c0);
    // 2) triangular causal MA
    csh_tdiff<<<BT / 256, 256, 0, stream>>>(c0, td);

    // 3) layer 1: (16x4096)@(4096x2048) + bias, SiLU
    csh_gemm16_partial<<<gemmBlocks, 256, 0, stream>>>(phases, cW1, part, T_, HALF_);
    csh_reduce_act    <<<redBlocks, 256, 0, stream>>>(part, cb1, h1c, HALF_, 1, 0, 0);
    csh_gemm16_partial<<<gemmBlocks, 256, 0, stream>>>(td,     pW1, part, T_, HALF_);
    csh_reduce_act    <<<redBlocks, 256, 0, stream>>>(part, pb1, h1p, HALF_, 1, 0, 0);

    // 4) layer 2: (16x2048)@(2048x2048) + bias, tanh -> interleaved output
    csh_gemm16_partial<<<gemmBlocks, 256, 0, stream>>>(h1c, cW2, part, HALF_, HALF_);
    csh_reduce_act    <<<redBlocks, 256, 0, stream>>>(part, cb2, out, HALF_, 2, 0, 1);
    csh_gemm16_partial<<<gemmBlocks, 256, 0, stream>>>(h1p, pW2, part, HALF_, HALF_);
    csh_reduce_act    <<<redBlocks, 256, 0, stream>>>(part, pb2, out, HALF_, 2, 1, 1);
}
